// MultiHeadAttentionWithRelativePE_31619549233182
// MI455X (gfx1250) — matrix operand order repaired and checked by reading, not compiled, not run on hardware
//
#include <hip/hip_runtime.h>

// ---------------------------------------------------------------------------
// MHA with relative position embeddings, fused flash-attention, CDNA5 WMMA
// ---------------------------------------------------------------------------

typedef __attribute__((ext_vector_type(16))) __bf16 v16bf;
typedef __attribute__((ext_vector_type(8)))  __bf16 v8bf;
typedef __attribute__((ext_vector_type(8)))  float  v8f;

#define NHEADS 12
#define KD     64
#define CH     768
#define HW     4096
#define N3C    2304

#define WMMA_BF16(A, B, C) \
  __builtin_amdgcn_wmma_f32_16x16x32_bf16(false, (A), false, (B), (short)0, (C), false, false)

__device__ __forceinline__ unsigned short f2bf(float f) {
  unsigned int u = __float_as_uint(f);
  u += 0x7FFFu + ((u >> 16) & 1u);   // round-to-nearest-even
  return (unsigned short)(u >> 16);
}
__device__ __forceinline__ float bf2f(unsigned short h) {
  return __uint_as_float(((unsigned int)h) << 16);
}
__device__ __forceinline__ v8f vzero8() {
  v8f z; 
#pragma unroll
  for (int i = 0; i < 8; ++i) z[i] = 0.f;
  return z;
}
__device__ __forceinline__ v16bf cat8(v8bf lo, v8bf hi) {
  return __builtin_shufflevector(lo, hi, 0,1,2,3,4,5,6,7,8,9,10,11,12,13,14,15);
}
// A-operand (16-bit 16x32): per lane, K runs {c0..c0+7} and {c0+16..c0+23}
// (caller folds the lane-half offset 0/8 and k-chunk base into p)
__device__ __forceinline__ v16bf load_rowA(const unsigned short* p) {
  v8bf lo = *(const v8bf*)(p);
  v8bf hi = *(const v8bf*)(p + 16);
  return cat8(lo, hi);
}
// B-operand: per lane, K runs {c0..c0+7} and {c0+16..c0+23} (same as A)
__device__ __forceinline__ v16bf load_rowB(const unsigned short* p) {
  v8bf lo = *(const v8bf*)(p);
  v8bf hi = *(const v8bf*)(p + 16);
  return cat8(lo, hi);
}

// ------------------------------ prep kernels -------------------------------

__global__ void k_cvt_bf16(const float* __restrict__ in,
                           unsigned short* __restrict__ out, int n) {
  int i = blockIdx.x * blockDim.x + threadIdx.x;
  if (i < n) out[i] = f2bf(in[i]);
}

// out[n*K + k] = bf16(in[k*N + n])   (transpose K x N -> N x K)
__global__ void k_transpose_bf16(const float* __restrict__ in,
                                 unsigned short* __restrict__ out, int K, int N) {
  int i = blockIdx.x * blockDim.x + threadIdx.x;
  if (i >= K * N) return;
  int n = i / K, k = i - n * K;
  out[i] = f2bf(in[k * N + n]);
}

// ------------------------------ QKV GEMM -----------------------------------
// qkv = x(4096x768) @ W(768x2304) + b ; scatter into q/k (head-major rows)
// and V transposed [head][d][key] for the P@V B-operand.
__global__ void __launch_bounds__(128)
k_qkv_gemm(const unsigned short* __restrict__ xb,    // [4096][768] bf16
           const unsigned short* __restrict__ wT,    // [2304][768] bf16
           const float*          __restrict__ qkv_b, // [2304]
           unsigned short* __restrict__ qb,          // [12][4096][64]
           unsigned short* __restrict__ kb,          // [12][4096][64]
           unsigned short* __restrict__ vT)          // [12][64][4096]
{
  const int lane = threadIdx.x & 31;
  const int wv   = threadIdx.x >> 5;
  const int l16  = lane & 15;
  const int hiA  = (lane < 16) ? 0 : 8;
  const int hiB  = (lane < 16) ? 0 : 8;
  const int mbase = blockIdx.x * 16;
  const int nbase = blockIdx.y * 256 + wv * 64;

  v8f acc[4];
#pragma unroll
  for (int nt = 0; nt < 4; ++nt) acc[nt] = vzero8();

  const unsigned short* arow = xb + (size_t)(mbase + l16) * CH;
  for (int kk = 0; kk < CH; kk += 32) {
    v16bf A = load_rowA(arow + kk + hiA);
#pragma unroll
    for (int nt = 0; nt < 4; ++nt) {
      v16bf B = load_rowB(wT + (size_t)(nbase + nt * 16 + l16) * CH + kk + hiB);
      acc[nt] = WMMA_BF16(A, B, acc[nt]);
    }
  }

#pragma unroll
  for (int nt = 0; nt < 4; ++nt) {
    int col   = nbase + nt * 16 + l16;
    int which = col / CH;
    int rem   = col - which * CH;
    int head  = rem >> 6, d = rem & 63;
    float bias = qkv_b[col];
#pragma unroll
    for (int r = 0; r < 8; ++r) {
      int row = mbase + r + hiA;
      unsigned short bv = f2bf(acc[nt][r] + bias);
      if (which == 0)      qb[((size_t)head * HW + row) * KD + d] = bv;
      else if (which == 1) kb[((size_t)head * HW + row) * KD + d] = bv;
      else                 vT[((size_t)head * KD + d) * HW + row] = bv;
    }
  }
}

// ------------------------- relative position biases ------------------------
// relh[head][q][k1] = sum_c q[head][q][c] * rel_pos_h[qh-k1+63][c]  (sel=0)
// relw[head][q][k2] = sum_c q[head][q][c] * rel_pos_w[qw-k2+63][c]  (sel=1)
__global__ void k_rel(const unsigned short* __restrict__ qb,
                      const float* __restrict__ rel_pos_h,
                      const float* __restrict__ rel_pos_w,
                      float* __restrict__ relh, float* __restrict__ relw)
{
  int id = blockIdx.x * blockDim.x + threadIdx.x;  // 2*12*4096*64 total
  int k = id & 63;
  int u = id >> 6;
  int qpos = u & (HW - 1);
  u >>= 12;
  int head = u % NHEADS;
  int sel  = u / NHEADS;

  int qh = qpos >> 6, qw = qpos & 63;
  const unsigned short* qrow = qb + ((size_t)head * HW + qpos) * KD;
  int ridx = ((sel == 0) ? qh : qw) - k + 63;
  const float* rp = ((sel == 0) ? rel_pos_h : rel_pos_w) + (size_t)ridx * KD;

  float s = 0.f;
#pragma unroll 8
  for (int c = 0; c < KD; ++c) s += bf2f(qrow[c]) * rp[c];

  float* out = (sel == 0) ? relh : relw;
  out[((size_t)head * HW + qpos) * 64 + k] = s;
}

// --------------------------- fused flash attention -------------------------
// Per wave: 16 query rows. Loop keys in chunks of 32.
// S = (Q*scale)K^T + rel_h + rel_w ; online softmax ; O += P @ V.
__global__ void __launch_bounds__(128)
k_flash_attn(const unsigned short* __restrict__ qb,   // [12][4096][64]
             const unsigned short* __restrict__ kb,   // [12][4096][64]
             const unsigned short* __restrict__ vT,   // [12][64][4096]
             const float* __restrict__ relh,          // [12][4096][64]
             const float* __restrict__ relw,          // [12][4096][64]
             unsigned short* __restrict__ attn_out)   // [4096][768] bf16
{
  __shared__ unsigned short pbuf[4 * 16 * 40];        // per-wave P bounce, stride 40
  const int lane = threadIdx.x & 31;
  const int wv   = threadIdx.x >> 5;
  const int l16  = lane & 15;
  const int hiA  = (lane < 16) ? 0 : 8;
  const int hiB  = (lane < 16) ? 0 : 8;
  const int head = blockIdx.y;
  const int qbase = blockIdx.x * 64 + wv * 16;
  const float scale = 0.125f;                         // 64^-0.5

  const unsigned short* kh = kb + (size_t)head * HW * KD;
  const unsigned short* vh = vT + (size_t)head * KD * HW;
  const unsigned short* qrowp = qb + ((size_t)head * HW + qbase + l16) * KD;

  v16bf qA0 = load_rowA(qrowp + 0  + hiA);            // channels 0..31
  v16bf qA1 = load_rowA(qrowp + 32 + hiA);            // channels 32..63

  float mstat[8], lstat[8];
  v8f o[4];
#pragma unroll
  for (int dt = 0; dt < 4; ++dt) o[dt] = vzero8();
#pragma unroll
  for (int r = 0; r < 8; ++r) { mstat[r] = -3.0e38f; lstat[r] = 0.f; }

  const float* rh[8];
  const float* rw[8];
#pragma unroll
  for (int r = 0; r < 8; ++r) {
    int qrow = qbase + r + hiA;
    rh[r] = relh + ((size_t)head * HW + qrow) * 64;
    rw[r] = relw + ((size_t)head * HW + qrow) * 64;
  }
  unsigned short* pl = pbuf + wv * 640;

  for (int kc = 0; kc < HW; kc += 32) {
    const int keyh = kc >> 6;      // constant across a 32-key chunk
    const int kw0  = kc & 63;

    // ---- S = Q K^T (two 16-key subtiles, K=64 contraction) ----
    v8f s0 = vzero8(), s1 = vzero8();
    {
      v16bf B0 = load_rowB(kh + (size_t)(kc + l16) * KD + 0  + hiB);
      v16bf B1 = load_rowB(kh + (size_t)(kc + l16) * KD + 32 + hiB);
      s0 = WMMA_BF16(qA0, B0, s0);
      s0 = WMMA_BF16(qA1, B1, s0);
      v16bf B2 = load_rowB(kh + (size_t)(kc + 16 + l16) * KD + 0  + hiB);
      v16bf B3 = load_rowB(kh + (size_t)(kc + 16 + l16) * KD + 32 + hiB);
      s1 = WMMA_BF16(qA0, B2, s1);
      s1 = WMMA_BF16(qA1, B3, s1);
    }

    // ---- scale + relative-position bias, row maxima ----
    float rowm[8];
#pragma unroll
    for (int r = 0; r < 8; ++r) {
      float bh = rh[r][keyh];
      float a = s0[r] * scale + bh + rw[r][kw0 + l16];
      float b = s1[r] * scale + bh + rw[r][kw0 + 16 + l16];
      s0[r] = a; s1[r] = b;
      rowm[r] = fmaxf(a, b);
    }
#pragma unroll
    for (int mask = 1; mask < 16; mask <<= 1)
#pragma unroll
      for (int r = 0; r < 8; ++r)
        rowm[r] = fmaxf(rowm[r], __shfl_xor(rowm[r], mask, 32));

    // ---- online softmax update ----
    float rsum[8];
#pragma unroll
    for (int r = 0; r < 8; ++r) {
      float mnew  = fmaxf(mstat[r], rowm[r]);
      float alpha = __expf(mstat[r] - mnew);
      float p0 = __expf(s0[r] - mnew);
      float p1 = __expf(s1[r] - mnew);
      s0[r] = p0; s1[r] = p1;
      rsum[r] = p0 + p1;
      mstat[r] = mnew;
      lstat[r] *= alpha;
      o[0][r] *= alpha; o[1][r] *= alpha; o[2][r] *= alpha; o[3][r] *= alpha;
    }
#pragma unroll
    for (int mask = 1; mask < 16; mask <<= 1)
#pragma unroll
      for (int r = 0; r < 8; ++r)
        rsum[r] += __shfl_xor(rsum[r], mask, 32);
#pragma unroll
    for (int r = 0; r < 8; ++r) lstat[r] += rsum[r];

    // ---- P: C-layout f32 -> LDS bf16 -> A-layout reload ----
#pragma unroll
    for (int r = 0; r < 8; ++r) {
      int m = r + hiA;
      pl[m * 40 + l16]      = f2bf(s0[r]);
      pl[m * 40 + 16 + l16] = f2bf(s1[r]);
    }
    v16bf pA = load_rowA(pl + l16 * 40 + hiA);

    // ---- O += P @ V (d = 4 x 16) ----
#pragma unroll
    for (int dt = 0; dt < 4; ++dt) {
      v16bf vB = load_rowB(vh + (size_t)(dt * 16 + l16) * HW + kc + hiB);
      o[dt] = WMMA_BF16(pA, vB, o[dt]);
    }
  }

  // ---- normalize and emit bf16 for proj GEMM ----
#pragma unroll
  for (int dt = 0; dt < 4; ++dt) {
    int col = head * 64 + dt * 16 + l16;
#pragma unroll
    for (int r = 0; r < 8; ++r) {
      int qrow = qbase + r + hiA;
      attn_out[(size_t)qrow * CH + col] = f2bf(o[dt][r] / lstat[r]);
    }
  }
}

// ------------------------------ proj GEMM ----------------------------------
__global__ void __launch_bounds__(128)
k_proj_gemm(const unsigned short* __restrict__ ab,     // [4096][768] bf16
            const unsigned short* __restrict__ wT,     // [768][768]  bf16
            const float*          __restrict__ proj_b, // [768]
            float*                __restrict__ out)    // [4096][768] f32
{
  const int lane = threadIdx.x & 31;
  const int wv   = threadIdx.x >> 5;
  const int l16  = lane & 15;
  const int hiA  = (lane < 16) ? 0 : 8;
  const int hiB  = (lane < 16) ? 0 : 8;
  const int mbase = blockIdx.x * 16;
  const int nbase = blockIdx.y * 256 + wv * 64;

  v8f acc[4];
#pragma unroll
  for (int nt = 0; nt < 4; ++nt) acc[nt] = vzero8();

  const unsigned short* arow = ab + (size_t)(mbase + l16) * CH;
  for (int kk = 0; kk < CH; kk += 32) {
    v16bf A = load_rowA(arow + kk + hiA);
#pragma unroll
    for (int nt = 0; nt < 4; ++nt) {
      v16bf B = load_rowB(wT + (size_t)(nbase + nt * 16 + l16) * CH + kk + hiB);
      acc[nt] = WMMA_BF16(A, B, acc[nt]);
    }
  }

#pragma unroll
  for (int nt = 0; nt < 4; ++nt) {
    int col = nbase + nt * 16 + l16;
    float b = proj_b[col];
#pragma unroll
    for (int r = 0; r < 8; ++r)
      out[(size_t)(mbase + r + hiA) * CH + col] = acc[nt][r] + b;
  }
}

// ------------------------------- launcher ----------------------------------
extern "C" void kernel_launch(void* const* d_in, const int* in_sizes, int n_in,
                              void* d_out, int out_size, void* d_ws, size_t ws_size,
                              hipStream_t stream)
{
  (void)in_sizes; (void)n_in; (void)out_size; (void)ws_size;
  const float* x      = (const float*)d_in[0];
  const float* qkv_w  = (const float*)d_in[1];
  const float* qkv_b  = (const float*)d_in[2];
  const float* proj_w = (const float*)d_in[3];
  const float* proj_b = (const float*)d_in[4];
  const float* rph    = (const float*)d_in[5];
  const float* rpw    = (const float*)d_in[6];
  float* out = (float*)d_out;

  char* ws = (char*)d_ws;
  size_t off = 0;
  auto alloc = [&](size_t bytes) -> char* {
    char* p = ws + off;
    off = (off + bytes + 255) & ~(size_t)255;
    return p;
  };
  unsigned short* xb  = (unsigned short*)alloc((size_t)HW * CH * 2);
  unsigned short* wqT = (unsigned short*)alloc((size_t)N3C * CH * 2);
  unsigned short* wpT = (unsigned short*)alloc((size_t)CH * CH * 2);
  unsigned short* qb  = (unsigned short*)alloc((size_t)NHEADS * HW * KD * 2);
  unsigned short* kb  = (unsigned short*)alloc((size_t)NHEADS * HW * KD * 2);
  unsigned short* vT  = (unsigned short*)alloc((size_t)NHEADS * HW * KD * 2);
  float* relh = (float*)alloc((size_t)NHEADS * HW * 64 * 4);
  float* relw = (float*)alloc((size_t)NHEADS * HW * 64 * 4);
  unsigned short* ab  = (unsigned short*)alloc((size_t)HW * CH * 2);

  k_cvt_bf16<<<(HW * CH + 255) / 256, 256, 0, stream>>>(x, xb, HW * CH);
  k_transpose_bf16<<<(CH * N3C + 255) / 256, 256, 0, stream>>>(qkv_w, wqT, CH, N3C);
  k_transpose_bf16<<<(CH * CH + 255) / 256, 256, 0, stream>>>(proj_w, wpT, CH, CH);

  k_qkv_gemm<<<dim3(HW / 16, N3C / 256), 128, 0, stream>>>(xb, wqT, qkv_b, qb, kb, vT);

  k_rel<<<(2 * NHEADS * HW * 64) / 256, 256, 0, stream>>>(qb, rph, rpw, relh, relw);

  k_flash_attn<<<dim3(HW / 64, NHEADS), 128, 0, stream>>>(qb, kb, vT, relh, relw, ab);

  k_proj_gemm<<<dim3(HW / 16, CH / 256), 128, 0, stream>>>(ab, wpT, proj_b, out);
}
